// Qwen3VLAttention_51848845197681
// MI455X (gfx1250) — compile-verified, matrix operand order checked by reading, not run on hardware
//
#include <hip/hip_runtime.h>
#include <hip/hip_bf16.h>
#include <stdint.h>

typedef __attribute__((ext_vector_type(16))) __bf16   v16bf;
typedef __attribute__((ext_vector_type(8)))  __bf16   v8bf;
typedef __attribute__((ext_vector_type(8)))  float    v8f;
typedef __attribute__((ext_vector_type(4)))  uint32_t u32x4;
typedef __attribute__((ext_vector_type(8)))  uint32_t u32x8;

static constexpr int SEQ = 4096;
static constexpr int HID = 2048;
static constexpr int NH  = 16;
static constexpr int NKV = 8;
static constexpr int HD  = 128;
static constexpr int QKVW = NH*HD + 2*NKV*HD;   // 4096
static constexpr float ATTN_SCALE = 0.08838834764831845f; // 128^-0.5

// ---- helpers -------------------------------------------------------------

__device__ __forceinline__ __bf16 f2bf(float f) {
  union { float f; uint32_t u; } x; x.f = f;
  uint32_t r = x.u + 0x7FFFu + ((x.u >> 16) & 1u);   // round-to-nearest-even
  uint16_t h = (uint16_t)(r >> 16);
  return __builtin_bit_cast(__bf16, h);
}

// Load one 16x16x32 bf16 A/B fragment for this lane.
// p points at (row-or-col base) in LDS; per ISA 7.12.2, lane's 16 bf16 are
// K = {half*8 .. half*8+7} and {16+half*8 .. 16+half*8+7}.
__device__ __forceinline__ v16bf ld_frag(const __bf16* p, int half) {
  v16bf f;
  *(v8bf*)&f       = *(const v8bf*)(p + half * 8);
  *((v8bf*)&f + 1) = *(const v8bf*)(p + 16 + half * 8);
  return f;
}

__device__ __forceinline__ v8f wmma_bf16(v16bf a, v16bf b, v8f c) {
  return __builtin_amdgcn_wmma_f32_16x16x32_bf16(false, a, false, b, (short)0, c,
                                                 false, false);
}

// TDM: DMA a 2D bf16 tile (tile_w elems x tile_h rows, row stride
// stride_elems) from global memory into LDS at lds_off, inserting (pad_amt+1)
// dwords of LDS padding every (1 << (pad_int+1)) dwords (per D# encoding).
// Issued by one wave; completion tracked with TENSORcnt.
__device__ __forceinline__ void tdm_load_2d(const void* gaddr, uint32_t lds_off,
                                            uint32_t tile_w, uint32_t tile_h,
                                            uint32_t stride_elems,
                                            uint32_t pad_int, uint32_t pad_amt) {
  uint64_t ga = (uint64_t)(uintptr_t)gaddr;
  u32x4 g0;
  g0[0] = 1u;                                    // count=1, user descriptor
  g0[1] = lds_off;                               // lds_addr
  g0[2] = (uint32_t)ga;                          // global_addr[31:0]
  g0[3] = (uint32_t)((ga >> 32) & 0x1FFFFFFu)    // global_addr[56:32]
        | (2u << 30);                            // type = 2 ("image")
  u32x8 g1;
  g1[0] = (1u << 16)                             // data_size = 2 bytes
        | (1u << 20)                             // pad_enable
        | (pad_int << 22) | (pad_amt << 25);
  g1[1] = (tile_w & 0xFFFFu) << 16;              // tensor_dim0[15:0]
  g1[2] = ((tile_w >> 16) & 0xFFFFu)             // tensor_dim0[31:16]
        | ((tile_h & 0xFFFFu) << 16);            // tensor_dim1[15:0]
  g1[3] = ((tile_h >> 16) & 0xFFFFu)             // tensor_dim1[31:16]
        | ((tile_w & 0xFFFFu) << 16);            // tile_dim0
  g1[4] = tile_h & 0xFFFFu;                      // tile_dim1 (tile_dim2 = 0)
  g1[5] = stride_elems;                          // tensor_dim0_stride[31:0]
  g1[6] = 0;                                     // stride hi / dim1_stride lo
  g1[7] = 0;
  asm volatile("tensor_load_to_lds %0, %1" :: "s"(g0), "s"(g1) : "memory");
}

// ---- Kernel 1/4: fp32 in, bf16 WMMA, fp32 out tiled GEMM ----------------
// C[M,N] = A[M,K] * B[K,N].  Block 256 thr (8 waves), tile 128x128, Kstep 32.

#define GKP 40   // LDS K-stride (elems): 80B, multiple of 16B

__global__ __launch_bounds__(256)
void gemm_f32_bf16wmma(const float* __restrict__ A, const float* __restrict__ B,
                       float* __restrict__ C, int M, int N, int K) {
  __shared__ __bf16 As[128 * GKP];
  __shared__ __bf16 Bs[128 * GKP];   // stored transposed: Bs[n][k]

  const int tid  = threadIdx.x;
  const int wave = tid >> 5, lane = tid & 31;
  const int half = lane >> 4, l15 = lane & 15;
  const int bm = blockIdx.y * 128, bn = blockIdx.x * 128;
  const int mw = wave >> 2;   // 0..1  (64-row band)
  const int nw = wave & 3;    // 0..3  (32-col band)

  v8f acc[4][2] = {};

  const int arow = tid >> 1, acol = (tid & 1) * 16;
  const int bk   = tid >> 3, bnb  = (tid & 7) * 16;
  const float* ag = A + (size_t)(bm + arow) * K + acol;
  const float* bg = B + (size_t)bk * N + bn + bnb;

  for (int k0 = 0; k0 < K; k0 += 32) {
    { // A tile 128x32: thread -> 16 floats, fp32 -> bf16 -> LDS
      __bf16* s = As + arow * GKP + acol;
#pragma unroll
      for (int i = 0; i < 16; i += 4) {
        float4 v = *(const float4*)(ag + i);
        s[i+0] = f2bf(v.x); s[i+1] = f2bf(v.y);
        s[i+2] = f2bf(v.z); s[i+3] = f2bf(v.w);
      }
      __builtin_prefetch(ag + 32, 0, 0);   // next K tile -> global_prefetch_b8
    }
    { // B tile 32x128, transpose into Bs[n][k]
#pragma unroll
      for (int i = 0; i < 16; i += 4) {
        float4 v = *(const float4*)(bg + i);
        Bs[(bnb+i+0)*GKP + bk] = f2bf(v.x);
        Bs[(bnb+i+1)*GKP + bk] = f2bf(v.y);
        Bs[(bnb+i+2)*GKP + bk] = f2bf(v.z);
        Bs[(bnb+i+3)*GKP + bk] = f2bf(v.w);
      }
      __builtin_prefetch(bg + (size_t)32 * N, 0, 0);
    }
    ag += 32;
    bg += (size_t)32 * N;
    __syncthreads();

    v16bf a[4];
#pragma unroll
    for (int i = 0; i < 4; ++i)
      a[i] = ld_frag(As + (mw*64 + i*16 + l15) * GKP, half);
#pragma unroll
    for (int j = 0; j < 2; ++j) {
      v16bf b = ld_frag(Bs + (nw*32 + j*16 + l15) * GKP, half);
#pragma unroll
      for (int i = 0; i < 4; ++i)
        acc[i][j] = wmma_bf16(a[i], b, acc[i][j]);
    }
    __syncthreads();
  }

  // epilogue: one base pointer, stride-N pointer increments only
  float* p0 = C + (size_t)(bm + mw*64 + 8*half) * N + bn + nw*32 + l15;
#pragma unroll
  for (int i = 0; i < 4; ++i) {
    float* p = p0 + (size_t)(i * 16) * N;
#pragma unroll
    for (int e = 0; e < 8; ++e) {
      p[0]  = acc[i][0][e];
      p[16] = acc[i][1][e];
      p += N;
    }
  }
}

// ---- Kernel 2: RMSNorm + mRoPE + bf16 repack ----------------------------
// One block (256 thr / 8 waves) per token. Wave w: copies V kv-head w into
// transposed layout, then norms+ropes head-groups {w, w+8, w+16} (0..15 = Q
// heads, 16..23 = K heads). Lane owns dims {lane, lane+32, lane+64, lane+96}
// so both RoPE pairs (d, d+64) stay in-lane.

__global__ __launch_bounds__(256)
void norm_rope_pack(const float* __restrict__ qkv, const int* __restrict__ pos,
                    const float* __restrict__ qw, const float* __restrict__ kw,
                    const float* __restrict__ cache,
                    __bf16* __restrict__ Qb,   // [NH][SEQ][HD]
                    __bf16* __restrict__ Kb,   // [NKV][SEQ][HD]
                    __bf16* __restrict__ Vt) { // [NKV][HD][SEQ]
  const int t = blockIdx.x;
  const int wave = threadIdx.x >> 5, lane = threadIdx.x & 31;
  const float* row = qkv + (size_t)t * QKVW;

  { // V copy -> transposed
    const float* v = row + NH*HD + NKV*HD + wave * HD;
#pragma unroll
    for (int i = 0; i < 4; ++i) {
      int d = lane + i * 32;
      Vt[((size_t)wave * HD + d) * SEQ + t] = f2bf(v[d]);
    }
  }

  // mRoPE cos/sin for dims i0=lane, i1=lane+32 (section select per ref)
  const int i0 = lane, i1 = lane + 32;
  const int s0 = i0 % 3;                       // i0 < 60 always
  const int s1 = (i1 < 60) ? (i1 % 3) : 0;
  const int p0 = pos[s0 * SEQ + t], p1 = pos[s1 * SEQ + t];
  const float c0 = cache[(size_t)p0 * HD + i0];
  const float n0 = cache[(size_t)p0 * HD + 64 + i0];
  const float c1 = cache[(size_t)p1 * HD + i1];
  const float n1 = cache[(size_t)p1 * HD + 64 + i1];

#pragma unroll
  for (int g = 0; g < 3; ++g) {
    const int hh  = wave + g * 8;        // 0..23
    const bool isq = hh < NH;
    const int h   = isq ? hh : hh - NH;
    const float* x = row + (isq ? h * HD : NH*HD + h * HD);
    const float* w = isq ? qw : kw;

    float x0 = x[i0], x1 = x[i1], x2 = x[i0 + 64], x3 = x[i1 + 64];
    float ss = x0*x0 + x1*x1 + x2*x2 + x3*x3;
#pragma unroll
    for (int m = 16; m >= 1; m >>= 1) ss += __shfl_xor(ss, m, 32);
    const float rs = rsqrtf(ss * (1.0f / 128.0f) + 1e-6f);
    x0 *= rs * w[i0]; x1 *= rs * w[i1]; x2 *= rs * w[i0+64]; x3 *= rs * w[i1+64];

    const float y0 = x0 * c0 - x2 * n0;
    const float y2 = x2 * c0 + x0 * n0;
    const float y1 = x1 * c1 - x3 * n1;
    const float y3 = x3 * c1 + x1 * n1;

    __bf16* dst = isq ? (Qb + ((size_t)h * SEQ + t) * HD)
                      : (Kb + ((size_t)h * SEQ + t) * HD);
    dst[i0] = f2bf(y0); dst[i1] = f2bf(y1);
    dst[i0 + 64] = f2bf(y2); dst[i1 + 64] = f2bf(y3);
  }
}

// ---- Kernel 3: causal flash attention (bf16 WMMA, TDM tile loads) -------
// 4 waves/block. Block = (query tile of 64, head). Wave w owns query rows
// [16w, 16w+16). Q/K/V tiles are DMA'd into LDS by the Tensor Data Mover
// (wave 0 issues, TENSORcnt-tracked, workgroup barrier publishes). The D#
// pad feature writes the bank-conflict padding during the DMA itself.
// Online-softmax stats (m, l, alpha) live as v8f in the WMMA C-layout
// (row = elem + 8*half), so row reductions are shfl_xor masks 1..8.

#define DPQ 136   // LDS dim-stride for Q/K: 64 dwords data + 4 dwords pad
#define SPK 72    // LDS key-stride for V/P: 32 dwords data + 4 dwords pad

__global__ __launch_bounds__(128)
void flash_attn_wmma(const __bf16* __restrict__ Qb, const __bf16* __restrict__ Kb,
                     const __bf16* __restrict__ Vt, float* __restrict__ Oout) {
  __shared__ __bf16 Qs[64 * DPQ];
  __shared__ __bf16 Ks[64 * DPQ];
  __shared__ __bf16 Vs[HD * SPK];       // Vs[d][key]
  __shared__ __bf16 Ps[4 * 16 * SPK];   // per-wave P (C->A relayout)

  const int qt = blockIdx.x, h = blockIdx.y;
  const int kvh = h >> 1;               // rep = NH/NKV = 2
  const int tid = threadIdx.x;
  const int wave = tid >> 5, lane = tid & 31;
  const int half = lane >> 4, l15 = lane & 15;

  if (wave == 0) {  // Q tile 64x128: TDM, pad every 64 dwords by 4 dwords
    tdm_load_2d(Qb + ((size_t)h * SEQ + qt * 64) * HD,
                (uint32_t)(uintptr_t)Qs, HD, 64, HD, 5, 3);
  }

  v8f m, l = {}, O[8] = {};
#pragma unroll
  for (int e = 0; e < 8; ++e) m[e] = -3.0e38f;

  for (int kb = 0; kb <= qt; ++kb) {
    __syncthreads();   // prev iter done reading Ks/Vs
    if (wave == 0) {
      // K block 64 rows x 128 dims
      tdm_load_2d(Kb + ((size_t)kvh * SEQ + kb * 64) * HD,
                  (uint32_t)(uintptr_t)Ks, HD, 64, HD, 5, 3);
      // V block (transposed layout): 128 rows x 64 keys, row stride SEQ;
      // pad every 32 dwords by 4 dwords
      tdm_load_2d(Vt + (size_t)kvh * HD * SEQ + kb * 64,
                  (uint32_t)(uintptr_t)Vs, 64, HD, SEQ, 4, 3);
      __builtin_amdgcn_s_wait_tensorcnt(0);   // also covers Q on iter 0
    }
    __syncthreads();

    // scores: 16 queries x 64 keys, inner dim 128 (4 K-chunks of 32)
    v8f sc[4] = {};
#pragma unroll
    for (int kc = 0; kc < 4; ++kc) {
      v16bf a = ld_frag(Qs + (wave*16 + l15) * DPQ + kc * 32, half);
#pragma unroll
      for (int sub = 0; sub < 4; ++sub) {
        v16bf b = ld_frag(Ks + (sub*16 + l15) * DPQ + kc * 32, half);
        sc[sub] = wmma_bf16(a, b, sc[sub]);
      }
    }

    const bool diag = (kb == qt);
#pragma unroll
    for (int sub = 0; sub < 4; ++sub)
#pragma unroll
      for (int e = 0; e < 8; ++e) {
        float v = sc[sub][e] * ATTN_SCALE;
        if (diag) {
          int r = wave*16 + e + 8*half;
          int c = sub*16 + l15;
          if (c > r) v = -3.0e38f;
        }
        sc[sub][e] = v;
      }

    // row max over the key block (subtiles, then 16-lane halves)
    v8f mb = sc[0];
#pragma unroll
    for (int sub = 1; sub < 4; ++sub)
#pragma unroll
      for (int e = 0; e < 8; ++e) mb[e] = fmaxf(mb[e], sc[sub][e]);
#pragma unroll
    for (int mk = 1; mk < 16; mk <<= 1)
#pragma unroll
      for (int e = 0; e < 8; ++e)
        mb[e] = fmaxf(mb[e], __shfl_xor(mb[e], mk, 32));

    v8f mnew, alpha, rsum = {};
#pragma unroll
    for (int e = 0; e < 8; ++e) {
      mnew[e]  = fmaxf(m[e], mb[e]);
      alpha[e] = __expf(m[e] - mnew[e]);
      m[e]     = mnew[e];
    }
#pragma unroll
    for (int sub = 0; sub < 4; ++sub)
#pragma unroll
      for (int e = 0; e < 8; ++e) {
        float p = __expf(sc[sub][e] - mnew[e]);
        sc[sub][e] = p;
        rsum[e] += p;
      }
#pragma unroll
    for (int mk = 1; mk < 16; mk <<= 1)
#pragma unroll
      for (int e = 0; e < 8; ++e) rsum[e] += __shfl_xor(rsum[e], mk, 32);
#pragma unroll
    for (int e = 0; e < 8; ++e) l[e] = l[e] * alpha[e] + rsum[e];
#pragma unroll
    for (int o = 0; o < 8; ++o)
#pragma unroll
      for (int e = 0; e < 8; ++e) O[o][e] *= alpha[e];

    // P: C-layout -> A-layout via wave-private LDS
    __bf16* pw = Ps + wave * 16 * SPK;
#pragma unroll
    for (int sub = 0; sub < 4; ++sub)
#pragma unroll
      for (int e = 0; e < 8; ++e)
        pw[(e + 8*half) * SPK + sub*16 + l15] = f2bf(sc[sub][e]);

    // O += P @ V  (inner dim 64 keys = 2 K-chunks)
#pragma unroll
    for (int kc = 0; kc < 2; ++kc) {
      v16bf a = ld_frag(pw + l15 * SPK + kc * 32, half);
#pragma unroll
      for (int sub = 0; sub < 8; ++sub) {
        v16bf b = ld_frag(Vs + (sub*16 + l15) * SPK + kc * 32, half);
        O[sub] = wmma_bf16(a, b, O[sub]);
      }
    }
  }

  // normalize and write attn output [SEQ][NH*HD] fp32 (const 2048 stride)
  float* p0 = Oout + (size_t)(qt*64 + wave*16 + 8*half) * (NH*HD) + h*HD + l15;
#pragma unroll
  for (int e = 0; e < 8; ++e) {
    const float inv = 1.0f / l[e];
#pragma unroll
    for (int sub = 0; sub < 8; ++sub)
      p0[sub * 16] = O[sub][e] * inv;
    p0 += NH * HD;
  }
}

// ---- host ---------------------------------------------------------------

extern "C" void kernel_launch(void* const* d_in, const int* in_sizes, int n_in,
                              void* d_out, int out_size, void* d_ws, size_t ws_size,
                              hipStream_t stream) {
  (void)in_sizes; (void)n_in; (void)out_size; (void)ws_size;
  const float* hs    = (const float*)d_in[0];
  const int*   pos   = (const int*)d_in[1];
  const float* w_qkv = (const float*)d_in[2];
  const float* w_o   = (const float*)d_in[3];
  const float* qw    = (const float*)d_in[4];
  const float* kw    = (const float*)d_in[5];
  const float* cache = (const float*)d_in[6];
  float* out = (float*)d_out;

  char* ws = (char*)d_ws;
  size_t off = 0;
  float*  qkv  = (float*)(ws + off);  off += (size_t)SEQ * QKVW * 4;      // 64 MB
  __bf16* Qb   = (__bf16*)(ws + off); off += (size_t)NH  * SEQ * HD * 2;  // 16 MB
  __bf16* Kb   = (__bf16*)(ws + off); off += (size_t)NKV * SEQ * HD * 2;  //  8 MB
  __bf16* Vt   = (__bf16*)(ws + off); off += (size_t)NKV * HD * SEQ * 2;  //  8 MB
  float*  attn = (float*)(ws + off);                                      // 32 MB

  dim3 g1(QKVW / 128, SEQ / 128);   // (32, 32)
  gemm_f32_bf16wmma<<<g1, 256, 0, stream>>>(hs, w_qkv, qkv, SEQ, QKVW, HID);

  norm_rope_pack<<<SEQ, 256, 0, stream>>>(qkv, pos, qw, kw, cache, Qb, Kb, Vt);

  dim3 g3(SEQ / 64, NH);            // (64, 16)
  flash_attn_wmma<<<g3, 128, 0, stream>>>(Qb, Kb, Vt, attn);

  dim3 g4(HID / 128, SEQ / 128);    // (16, 32)
  gemm_f32_bf16wmma<<<g4, 256, 0, stream>>>(attn, w_o, out, SEQ, HID, NH * HD);
}
